// PCTransformer_28982439313704
// MI455X (gfx1250) — compile-verified, matrix operand order checked by reading, not run on hardware
//
#include <hip/hip_runtime.h>
#include <stdint.h>

// ---------------------------------------------------------------------------
// MI455X (gfx1250) implementation of the VN point-cloud transformer block.
// All GEMMs + attention use v_wmma_f32_16x16x32_bf16 (fp32 accumulate).
// Edge-conv folded algebraically: mean_k We*[nb-x; x] = We_a*(mean nb - x) + We_b*x.
// Attention is flash-style (no 134MB score tensor).
// GEMM weight tiles staged to LDS via global_load_async_to_lds_b128.
// ---------------------------------------------------------------------------

typedef unsigned short u16;
typedef __attribute__((ext_vector_type(16))) __bf16 v16bf;
typedef __attribute__((ext_vector_type(8)))  __bf16 v8bf;
typedef __attribute__((ext_vector_type(8)))  float  v8f;
typedef __attribute__((ext_vector_type(8)))  u16    v8u;

#define EPSF   1e-6f
#define SLOPEF 0.2f

static __device__ __forceinline__ u16 f2bf(float f) {
  union { float f; unsigned u; } x; x.f = f;
  unsigned r = x.u + 0x7FFFu + ((x.u >> 16) & 1u);   // round-to-nearest-even
  return (u16)(r >> 16);
}
static __device__ __forceinline__ float bf2f(u16 h) {
  union { unsigned u; float f; } x; x.u = ((unsigned)h) << 16;
  return x.f;
}
static __device__ __forceinline__ v16bf cat8(v8bf lo, v8bf hi) {
  v16bf o;
#pragma unroll
  for (int i = 0; i < 8; ++i) { o[i] = lo[i]; o[i + 8] = hi[i]; }
  return o;
}
// 16x32 bf16 WMMA fragment from row-major [rows x ld] matrix (rows = M/N index,
// columns = K index). Lane l -> row l&15; half (l>>4) selects K offsets per the
// CDNA5 16-bit A-matrix VGPR layout (K = {half*8+0..7, 16+half*8+0..7}).
static __device__ __forceinline__ v16bf load_frag(const u16* base, long row0, int ld,
                                                  int k0, int lane) {
  int r = lane & 15, half = lane >> 4;
  const u16* p = base + (row0 + r) * (long)ld + k0 + half * 8;
  v8bf lo = *(const v8bf*)p;
  v8bf hi = *(const v8bf*)(p + 16);
  return cat8(lo, hi);
}
static __device__ __forceinline__ float rmax16(float v) {
#pragma unroll
  for (int m = 1; m < 16; m <<= 1) v = fmaxf(v, __shfl_xor(v, m, 32));
  return v;
}
static __device__ __forceinline__ float rsum16(float v) {
#pragma unroll
  for (int m = 1; m < 16; m <<= 1) v += __shfl_xor(v, m, 32);
  return v;
}

// ---------------------------------------------------------------------------
// f32 -> bf16 weight conversion
// ---------------------------------------------------------------------------
__global__ void __launch_bounds__(256) cvt_kernel(const float* __restrict__ s,
                                                  u16* __restrict__ d, int n) {
  int i = blockIdx.x * 256 + threadIdx.x;
  if (i < n) d[i] = f2bf(s[i]);
}

// ---------------------------------------------------------------------------
// vec_layernorm: x f32 [B,C,3,N] -> transposed bf16 [B, 3N, C] (rows = d*N+n)
// ---------------------------------------------------------------------------
__global__ void __launch_bounds__(256) ln_kernel(const float* __restrict__ x,
                                                 u16* __restrict__ xt, int C, int N) {
  int b = blockIdx.y, n = blockIdx.x, c = threadIdx.x;
  long base = ((long)(b * C + c) * 3) * N + n;
  float v0 = x[base], v1 = x[base + N], v2 = x[base + 2 * N];
  float nv = sqrtf(v0 * v0 + v1 * v1 + v2 * v2 + EPSF);
  __shared__ float red[256];
  red[c] = nv; __syncthreads();
  for (int s = 128; s > 0; s >>= 1) { if (c < s) red[c] += red[c + s]; __syncthreads(); }
  float mu = red[0] / (float)C; __syncthreads();
  float dv = nv - mu;
  red[c] = dv * dv; __syncthreads();
  for (int s = 128; s > 0; s >>= 1) { if (c < s) red[c] += red[c + s]; __syncthreads(); }
  float var = red[0] / (float)C;
  float s = (nv - mu) * rsqrtf(var + EPSF) / nv;
  long ob = (long)b * 3 * N * C;
  xt[ob + ((long)0 * N + n) * C + c] = f2bf(v0 * s);
  xt[ob + ((long)1 * N + n) * C + c] = f2bf(v1 * s);
  xt[ob + ((long)2 * N + n) * C + c] = f2bf(v2 * s);
}

// ---------------------------------------------------------------------------
// gather-mean for kNN edge conv:
//   cat[b, (d,n), 0:C]   = mean_k xv[:, :, idx[n,k]] - xq[:, :, n]
//   cat[b, (d,n), C:2C]  = xq[:, :, n]
// both xq/xv in transposed bf16 layout [B, 3N, C].
// ---------------------------------------------------------------------------
__global__ void __launch_bounds__(256) gather_kernel(const u16* __restrict__ xqT,
                                                     const u16* __restrict__ xvT,
                                                     const int* __restrict__ idx,
                                                     u16* __restrict__ cat,
                                                     int C, int N, int K) {
  int b = blockIdx.y, n = blockIdx.x, c = threadIdx.x;
  __shared__ int sidx[32];
  if (c < K) sidx[c] = idx[((long)b * N + n) * K + c];
  __syncthreads();
  long xb = (long)b * 3 * N * C;
  long cb = (long)b * 3 * N * (2 * C);
  float invK = 1.0f / (float)K;
#pragma unroll
  for (int d = 0; d < 3; ++d) {
    float acc = 0.f;
    for (int k = 0; k < K; ++k)
      acc += bf2f(xvT[xb + ((long)d * N + sidx[k]) * C + c]);
    float ctr = bf2f(xqT[xb + ((long)d * N + n) * C + c]);
    long row = cb + ((long)d * N + n) * (2 * C);
    cat[row + c]     = f2bf(acc * invK - ctr);
    cat[row + C + c] = f2bf(ctr);
  }
}

// ---------------------------------------------------------------------------
// GEMM: Y[o,m] = sum_k W[o,k] * Xt[m,k]   (per batch, weights shared)
//  W  : bf16 row-major [O, K]
//  Xt : bf16 row-major [M, K]  (batch stride M*K)
//  MODE 0: Y bf16 natural [O, M]
//  MODE 1: Y bf16 transposed [M, ldy], columns written at col_off (for concat)
//  MODE 2: Y f32 natural [O, M], residual add (Y += result)
// Block = 4 waves, 64x64 output tile; weight tile (64x32) staged to LDS with
// CDNA5 async loads. B fragments load straight from Xt (two b128 per lane).
// ---------------------------------------------------------------------------
template <int MODE>
__global__ void __launch_bounds__(128) gemm_kernel(const u16* __restrict__ W,
                                                   const u16* __restrict__ Xt,
                                                   void* __restrict__ Y,
                                                   int M, int Kd, int ldy,
                                                   int col_off, long yBatch) {
  __shared__ __align__(16) u16 smW[64 * 32];   // single LDS object -> offset 0
  int b  = blockIdx.z;
  int m0 = blockIdx.x * 64;
  int o0 = blockIdx.y * 64;
  int t  = threadIdx.x;
  int w  = t >> 5;
  int lane = t & 31;
  int r15 = lane & 15, half = lane >> 4;
  const u16* Xb = Xt + (long)b * M * Kd;

  v8f acc[4];
#pragma unroll
  for (int i = 0; i < 4; ++i)
#pragma unroll
    for (int j = 0; j < 8; ++j) acc[i][j] = 0.f;

  for (int k0 = 0; k0 < Kd; k0 += 32) {
    __syncthreads();
    // Async-stage W tile [64 x 32] bf16 into LDS (smW assumed at LDS offset 0).
#pragma unroll
    for (int c = 0; c < 2; ++c) {
      int chunk = t + c * 128;                // 256 x 16B = 4KB
      int row = chunk >> 2, q = chunk & 3;
      const u16* gp = W + (long)(o0 + row) * Kd + k0 + q * 8;
      unsigned lds_off = (unsigned)chunk * 16u;
      unsigned long long ga = (unsigned long long)(uintptr_t)gp;
      asm volatile("global_load_async_to_lds_b128 %0, %1, off"
                   :: "v"(lds_off), "v"(ga) : "memory");
    }
    if (k0 + 32 < Kd)
      __builtin_prefetch(W + (long)o0 * Kd + k0 + 32, 0, 1);   // global_prefetch_b8
    asm volatile("s_wait_asynccnt 0" ::: "memory");
    __syncthreads();

    // A fragment from LDS (this wave's 16 weight rows)
    const u16* pa = &smW[(w * 16 + r15) * 32 + half * 8];
    v16bf a = cat8(*(const v8bf*)pa, *(const v8bf*)(pa + 16));
#pragma unroll
    for (int ct = 0; ct < 4; ++ct) {
      v16bf bf = load_frag(Xb, m0 + ct * 16, Kd, k0, lane);
      acc[ct] = __builtin_amdgcn_wmma_f32_16x16x32_bf16(
          false, a, false, bf, (short)0, acc[ct], false, false);
    }
  }

  if (MODE == 1) {
    u16* Yp = (u16*)Y;
#pragma unroll
    for (int ct = 0; ct < 4; ++ct) {
      long mrow = m0 + ct * 16 + r15;
      u16* yp = Yp + b * yBatch + mrow * (long)ldy + col_off + o0 + w * 16 + half * 8;
      v8u pk;
#pragma unroll
      for (int i = 0; i < 8; ++i) pk[i] = f2bf(acc[ct][i]);
      *(v8u*)yp = pk;
    }
  } else if (MODE == 0) {
    u16* Yp = (u16*)Y;
#pragma unroll
    for (int ct = 0; ct < 4; ++ct) {
      int m = m0 + ct * 16 + r15;
#pragma unroll
      for (int i = 0; i < 8; ++i) {
        int o = o0 + w * 16 + i + half * 8;
        Yp[b * yBatch + (long)o * M + m] = f2bf(acc[ct][i]);
      }
    }
  } else {
    float* Yp = (float*)Y;
#pragma unroll
    for (int ct = 0; ct < 4; ++ct) {
      int m = m0 + ct * 16 + r15;
#pragma unroll
      for (int i = 0; i < 8; ++i) {
        int o = o0 + w * 16 + i + half * 8;
        long ix = b * yBatch + (long)o * M + m;
        Yp[ix] += acc[ct][i];                 // residual accumulate
      }
    }
  }
}

// ---------------------------------------------------------------------------
// Flash attention over (head_dim x 3) inner product.
//  qt, kt: bf16 [B, 3N, C]   (row = d*N+n, head h = cols h*hd..h*hd+hd)
//  vh    : bf16 [B, C, 3, N] (natural; B-fragment contiguous over nk)
//  ot    : bf16 [B, 3N, C]   (transposed, feeds Wo GEMM)
// One wave = 16 query rows; block = 4 waves; loop nk in 32-chunks.
// ---------------------------------------------------------------------------
__global__ void __launch_bounds__(128) flash_kernel(const u16* __restrict__ qt,
                                                    const u16* __restrict__ kt,
                                                    const u16* __restrict__ vh,
                                                    u16* __restrict__ ot,
                                                    int C, int N, int H) {
  __shared__ __align__(16) u16 smP[4][16 * 40];
  int bh = blockIdx.y;
  int b = bh / H, h = bh % H;
  int hd = C / H;                 // 64
  int hc0 = h * hd;
  int lane = threadIdx.x & 31, w = threadIdx.x >> 5;
  int r15 = lane & 15, half = lane >> 4;
  int nq0 = blockIdx.x * 64 + w * 16;
  long NE3 = (long)3 * N * C;
  const u16* qb = qt + (long)b * NE3;
  const u16* kb = kt + (long)b * NE3;
  const u16* vb = vh + (long)b * NE3;
  u16* ob = ot + (long)b * NE3;
  float scale = rsqrtf((float)(hd * 3));

  float mrow[8], lrow[8];
#pragma unroll
  for (int i = 0; i < 8; ++i) { mrow[i] = -1e30f; lrow[i] = 0.f; }
  v8f Oa[12];
#pragma unroll
  for (int j = 0; j < 12; ++j)
#pragma unroll
    for (int i = 0; i < 8; ++i) Oa[j][i] = 0.f;
  u16* myP = &smP[w][0];

  for (int nk0 = 0; nk0 < N; nk0 += 32) {
    v8f s0, s1;
#pragma unroll
    for (int i = 0; i < 8; ++i) { s0[i] = 0.f; s1[i] = 0.f; }
    // scores: contract over (d, c) = 3 x 64, chunks of 32
#pragma unroll
    for (int d = 0; d < 3; ++d) {
#pragma unroll
      for (int cc = 0; cc < 64; cc += 32) {
        v16bf a  = load_frag(qb, (long)d * N + nq0,      C, hc0 + cc, lane);
        v16bf k0f = load_frag(kb, (long)d * N + nk0,      C, hc0 + cc, lane);
        v16bf k1f = load_frag(kb, (long)d * N + nk0 + 16, C, hc0 + cc, lane);
        s0 = __builtin_amdgcn_wmma_f32_16x16x32_bf16(false, a, false, k0f, (short)0, s0, false, false);
        s1 = __builtin_amdgcn_wmma_f32_16x16x32_bf16(false, a, false, k1f, (short)0, s1, false, false);
      }
    }
    // online softmax; rows of the D tile live across 16-lane groups
    float alpha[8];
#pragma unroll
    for (int r = 0; r < 8; ++r) {
      float a0 = s0[r] * scale, a1 = s1[r] * scale;
      float mx = rmax16(fmaxf(a0, a1));
      float mn = fmaxf(mrow[r], mx);
      float al = __expf(mrow[r] - mn);
      float p0 = __expf(a0 - mn), p1 = __expf(a1 - mn);
      lrow[r] = lrow[r] * al + rsum16(p0 + p1);
      mrow[r] = mn;
      alpha[r] = al;
      myP[(r + 8 * half) * 40 + r15]      = f2bf(p0);
      myP[(r + 8 * half) * 40 + 16 + r15] = f2bf(p1);
    }
#pragma unroll
    for (int j = 0; j < 12; ++j)
#pragma unroll
      for (int r = 0; r < 8; ++r) Oa[j][r] *= alpha[r];
    asm volatile("s_wait_dscnt 0" ::: "memory");   // LDS in-order per wave
    v16bf pf = load_frag(myP, 0, 40, 0, lane);     // P as A fragment
    // O += P (16x32) x V (32x16) over the 12 (c,d) column tiles
#pragma unroll
    for (int j = 0; j < 12; ++j) {
      int jj = j * 16 + r15;
      int cl = jj / 3, dd = jj - cl * 3;
      const u16* p = vb + ((long)(hc0 + cl) * 3 + dd) * N + nk0 + half * 8;
      v16bf bf = cat8(*(const v8bf*)p, *(const v8bf*)(p + 16));
      Oa[j] = __builtin_amdgcn_wmma_f32_16x16x32_bf16(false, pf, false, bf, (short)0, Oa[j], false, false);
    }
  }
  // normalize and store transposed [3N, C]
#pragma unroll
  for (int j = 0; j < 12; ++j) {
    int jj = j * 16 + r15;
    int cl = jj / 3, dd = jj - cl * 3;
#pragma unroll
    for (int r = 0; r < 8; ++r) {
      int nq = nq0 + r + 8 * half;
      ob[((long)dd * N + nq) * C + hc0 + cl] = f2bf(Oa[j][r] / lrow[r]);
    }
  }
}

// ---------------------------------------------------------------------------
// Vector-Neuron LeakyReLU (elementwise over (b, ch, n); vectors over d=0..2)
//  hT, dT: bf16 [B, 3N, Cm]; result written in-place into dT.
// ---------------------------------------------------------------------------
__global__ void __launch_bounds__(256) leaky_kernel(const u16* __restrict__ hT,
                                                    u16* __restrict__ dT,
                                                    int Cm, int N) {
  long tid = (long)blockIdx.x * 256 + threadIdx.x;   // over B*N*Cm
  int ch = (int)(tid % Cm);
  long t2 = tid / Cm;
  int n = (int)(t2 % N);
  int b = (int)(t2 / N);
  long base = (long)b * 3 * N * Cm;
  float hv[3], dv[3];
#pragma unroll
  for (int d = 0; d < 3; ++d) {
    long ix = base + ((long)d * N + n) * Cm + ch;
    hv[d] = bf2f(hT[ix]);
    dv[d] = bf2f(dT[ix]);
  }
  float dot = hv[0] * dv[0] + hv[1] * dv[1] + hv[2] * dv[2];
  float dsq = dv[0] * dv[0] + dv[1] * dv[1] + dv[2] * dv[2] + EPSF;
  float f = (dot < 0.f) ? (dot / dsq) : 0.f;
#pragma unroll
  for (int d = 0; d < 3; ++d) {
    float xp = hv[d] - f * dv[d];
    float out = SLOPEF * hv[d] + (1.f - SLOPEF) * xp;
    dT[base + ((long)d * N + n) * Cm + ch] = f2bf(out);
  }
}

// ---------------------------------------------------------------------------
extern "C" void kernel_launch(void* const* d_in, const int* in_sizes, int n_in,
                              void* d_out, int out_size, void* d_ws, size_t ws_size,
                              hipStream_t stream) {
  const int B = 2, C = 256, N = 2048, K = 10, H = 4, Cm = 1024;
  const int M3 = 3 * N;
  const long NE = (long)B * C * 3 * N;

  const float* q = (const float*)d_in[0];
  const float* v = (const float*)d_in[1];
  const int* idxS = (const int*)d_in[19];
  const int* idxC = (const int*)d_in[20];

  u16* wsp = (u16*)d_ws;
  long off = 0;
  u16* nxt   = wsp + off; off += NE;                 // LN of x (transposed)
  u16* nvt   = wsp + off; off += NE;                 // LN of v (transposed)
  u16* qt    = wsp + off; off += NE;
  u16* kt    = wsp + off; off += NE;
  u16* vh    = wsp + off; off += NE;                 // natural layout
  u16* ot    = wsp + off; off += NE;                 // attention output (transposed)
  u16* catG  = wsp + off; off += (long)B * M3 * 2 * C;
  u16* catAG = wsp + off; off += (long)B * M3 * 2 * C;
  u16* hT    = wsp + off; off += (long)B * M3 * Cm;
  u16* dT    = wsp + off; off += (long)B * M3 * Cm;
  const int wsz[15] = {C * C, C * C, C * C, C * C, 2 * C * C, 2 * C * C,
                       C * C, C * C, C * C, C * C, 2 * C * C, 2 * C * C,
                       Cm * C, Cm * Cm, C * Cm};
  u16* wb[15];
  for (int i = 0; i < 15; ++i) { wb[i] = wsp + off; off += wsz[i]; }

  for (int i = 0; i < 15; ++i)
    cvt_kernel<<<(wsz[i] + 255) / 256, 256, 0, stream>>>((const float*)d_in[4 + i], wb[i], wsz[i]);

  hipMemcpyAsync(d_out, q, NE * sizeof(float), hipMemcpyDeviceToDevice, stream);
  float* x = (float*)d_out;

  dim3 gLN(N, B);
  dim3 gG(M3 / 64, C / 64, B);
  dim3 gGm(M3 / 64, Cm / 64, B);
  dim3 gF(N / 64, B * H);

  // ---------------- self attention block ----------------
  ln_kernel<<<gLN, 256, 0, stream>>>(x, nxt, C, N);
  gemm_kernel<1><<<gG, 128, 0, stream>>>(wb[0], nxt, qt, M3, C, C, 0, (long)M3 * C);
  gemm_kernel<1><<<gG, 128, 0, stream>>>(wb[1], nxt, kt, M3, C, C, 0, (long)M3 * C);
  gemm_kernel<0><<<gG, 128, 0, stream>>>(wb[2], nxt, vh, M3, C, 0, 0, (long)C * M3);
  flash_kernel<<<gF, 128, 0, stream>>>(qt, kt, vh, ot, C, N, H);
  gemm_kernel<1><<<gG, 128, 0, stream>>>(wb[3], ot, catAG, M3, C, 2 * C, 0, (long)M3 * 2 * C);
  gather_kernel<<<gLN, 256, 0, stream>>>(nxt, nxt, idxS, catG, C, N, K);
  gemm_kernel<1><<<gG, 128, 0, stream>>>(wb[4], catG, catAG, M3, 2 * C, 2 * C, C, (long)M3 * 2 * C);
  gemm_kernel<2><<<gG, 128, 0, stream>>>(wb[5], catAG, x, M3, 2 * C, 0, 0, (long)C * M3);

  // ---------------- cross attention block ----------------
  ln_kernel<<<gLN, 256, 0, stream>>>(x, nxt, C, N);
  ln_kernel<<<gLN, 256, 0, stream>>>(v, nvt, C, N);
  gemm_kernel<1><<<gG, 128, 0, stream>>>(wb[6], nxt, qt, M3, C, C, 0, (long)M3 * C);
  gemm_kernel<1><<<gG, 128, 0, stream>>>(wb[7], nvt, kt, M3, C, C, 0, (long)M3 * C);
  gemm_kernel<0><<<gG, 128, 0, stream>>>(wb[8], nvt, vh, M3, C, 0, 0, (long)C * M3);
  flash_kernel<<<gF, 128, 0, stream>>>(qt, kt, vh, ot, C, N, H);
  gemm_kernel<1><<<gG, 128, 0, stream>>>(wb[9], ot, catAG, M3, C, 2 * C, 0, (long)M3 * 2 * C);
  gather_kernel<<<gLN, 256, 0, stream>>>(nxt, nvt, idxC, catG, C, N, K);
  gemm_kernel<1><<<gG, 128, 0, stream>>>(wb[10], catG, catAG, M3, 2 * C, 2 * C, C, (long)M3 * 2 * C);
  gemm_kernel<2><<<gG, 128, 0, stream>>>(wb[11], catAG, x, M3, 2 * C, 0, 0, (long)C * M3);

  // ---------------- vector MLP block ----------------
  ln_kernel<<<gLN, 256, 0, stream>>>(x, nxt, C, N);
  gemm_kernel<1><<<gGm, 128, 0, stream>>>(wb[12], nxt, hT, M3, C, Cm, 0, (long)M3 * Cm);
  gemm_kernel<1><<<gGm, 128, 0, stream>>>(wb[13], hT, dT, M3, Cm, Cm, 0, (long)M3 * Cm);
  leaky_kernel<<<(int)(((long)B * N * Cm) / 256), 256, 0, stream>>>(hT, dT, Cm, N);
  gemm_kernel<2><<<gG, 128, 0, stream>>>(wb[14], dT, x, M3, Cm, 0, 0, (long)C * M3);
}
